// GCNDisc_46918222742366
// MI455X (gfx1250) — matrix operand order repaired and checked by reading, not compiled, not run on hardware
//
#include <hip/hip_runtime.h>
#include <hip/hip_bf16.h>
#include <math.h>

// ---------------- constants matching the reference ----------------
#define BATCH 64
#define NCROSS 32
#define NPIECE 128
#define NPTS 2048

typedef __attribute__((ext_vector_type(16))) _Float16 v16h;
typedef __attribute__((ext_vector_type(8)))  float    v8f;

__device__ __forceinline__ float leakyf(float x, float s) { return x > 0.f ? x : s * x; }

__device__ __forceinline__ void atomicMaxFloat(float* addr, float val) {
    unsigned int* ua = (unsigned int*)addr;
    unsigned int old = __float_as_uint(*addr);
    while (__uint_as_float(old) < val) {
        unsigned int prev = atomicCAS(ua, old, __float_as_uint(val));
        if (prev == old) break;
        old = prev;
    }
}

// =====================================================================
// WMMA GEMM: Y[rows x NOUT] = X[rows x FIN] @ W[FIN x NOUT]
// f16 inputs, f32 accumulate, v_wmma_f32_16x16x32_f16.
// Block = 256 threads = 8 waves; each wave computes a 16 x NOUT tile.
// Weights staged in LDS pre-swizzled into WMMA B-fragment layout so each
// lane reads its 16 halves as one contiguous 32B chunk (2x ds_load_b128).
// Requires nrows % 128 == 0 (true for all call sites: 262144 / 2048).
// =====================================================================
template <int FIN, int KPAD, int NOUT>
__global__ __launch_bounds__(256) void wmma_gemm_kernel(
    const float* __restrict__ X, const float* __restrict__ W, float* __restrict__ Y)
{
    constexpr int NT = NOUT / 16;
    constexpr int KSTEPS = KPAD / 32;
    // fragment layout: [(ks*NT + t)][lane][16 halves], contiguous per lane
    __shared__ __align__(16) _Float16 Wf[KSTEPS * NT * 32 * 16];

    int tid = threadIdx.x;
    for (int idx = tid; idx < KPAD * NOUT; idx += 256) {
        int k = idx / NOUT, n = idx % NOUT;
        int ks = k >> 5, kk = k & 31;
        int t = n >> 4, col = n & 15;
        int blane = col + ((kk & 8) << 1);     // lanes 0-15: K {0..7,16..23}, lanes 16-31: K {8..15,24..31}
        int j = (kk & 7) | ((kk & 16) >> 1);
        Wf[(((ks * NT + t) * 32) + blane) * 16 + j] =
            (k < FIN) ? (_Float16)W[k * NOUT + n] : (_Float16)0.f;
    }
    __syncthreads();

    int lane = tid & 31;
    int wave = tid >> 5;
    bool hi = lane >= 16;
    int lrow = lane & 15;
    size_t arow = (size_t)blockIdx.x * 128 + wave * 16 + lrow;  // A-row owned by this lane
    const int kb = hi ? 8 : 0;

    v8f acc[NT];
    #pragma unroll
    for (int t = 0; t < NT; t++) { v8f z = {}; acc[t] = z; }

    #pragma unroll
    for (int ks = 0; ks < KSTEPS; ks++) {
        const int k0 = ks * 32;
        // A fragment: lanes 0-15 hold K = {0..7,16..23}, lanes 16-31 hold K = {8..15,24..31}
        v16h a;
        if constexpr (FIN % 8 == 0) {
            const float4* p0 = (const float4*)(X + arow * FIN + k0 + kb);
            const float4* p1 = (const float4*)(X + arow * FIN + k0 + 16 + kb);
            float4 c0 = p0[0], c1 = p0[1];
            float4 c2 = p1[0], c3 = p1[1];
            a[0]  = (_Float16)c0.x; a[1]  = (_Float16)c0.y; a[2]  = (_Float16)c0.z; a[3]  = (_Float16)c0.w;
            a[4]  = (_Float16)c1.x; a[5]  = (_Float16)c1.y; a[6]  = (_Float16)c1.z; a[7]  = (_Float16)c1.w;
            a[8]  = (_Float16)c2.x; a[9]  = (_Float16)c2.y; a[10] = (_Float16)c2.z; a[11] = (_Float16)c2.w;
            a[12] = (_Float16)c3.x; a[13] = (_Float16)c3.y; a[14] = (_Float16)c3.z; a[15] = (_Float16)c3.w;
        } else {
            #pragma unroll
            for (int j = 0; j < 8; j++) {
                int k = k0 + kb + j;
                a[j] = (k < FIN) ? (_Float16)X[arow * FIN + k] : (_Float16)0.f;
            }
            #pragma unroll
            for (int j = 0; j < 8; j++) {
                int k = k0 + 16 + kb + j;
                a[8 + j] = (k < FIN) ? (_Float16)X[arow * FIN + k] : (_Float16)0.f;
            }
        }
        #pragma unroll
        for (int t = 0; t < NT; t++) {
            v16h b = *(const v16h*)&Wf[(((ks * NT + t) * 32) + lane) * 16];
            acc[t] = __builtin_amdgcn_wmma_f32_16x16x32_f16(
                false, a, false, b, (short)0, acc[t], false, false);
        }
    }

    // C/D layout: vgpr v -> row v (+8 for hi lanes), col = lane&15
    size_t rbase = (size_t)blockIdx.x * 128 + wave * 16 + (hi ? 8 : 0);
    float* yb = Y + rbase * NOUT + lrow;
    #pragma unroll
    for (int t = 0; t < NT; t++) {
        #pragma unroll
        for (int v = 0; v < 8; v++)
            yb[(size_t)v * NOUT + t * 16] = acc[t][v];
    }
}

// ---------------- degree / dinv ----------------
__global__ void fill_kernel(float* __restrict__ p, int n, float v) {
    int i = blockIdx.x * blockDim.x + threadIdx.x;
    if (i < n) p[i] = v;
}
__global__ void deg_kernel(const int* __restrict__ dst, float* __restrict__ deg, int E) {
    int e = blockIdx.x * blockDim.x + threadIdx.x;
    if (e < E) atomicAdd(&deg[dst[e]], 1.0f);
}
__global__ void rsqrt_kernel(float* __restrict__ p, int n) {
    int i = blockIdx.x * blockDim.x + threadIdx.x;
    if (i < n) p[i] = rsqrtf(p[i]);
}

// ---------------- GCN aggregation ----------------
// out[i][f] = bias[f] + dinv[i]^2 * xw[i][f]   (self loop, also initializes out)
__global__ void gcn_self_kernel(const float* __restrict__ xw, const float* __restrict__ dinv,
                                const float* __restrict__ bias, float* __restrict__ out,
                                int n, int F) {
    int idx = blockIdx.x * blockDim.x + threadIdx.x;
    if (idx >= n * F) return;
    int i = idx / F, f = idx % F;
    out[idx] = bias[f] + dinv[i] * dinv[i] * xw[idx];
}
// one wave per edge, lanes stride over features
__global__ __launch_bounds__(256) void gcn_edge_kernel(
    const float* __restrict__ xw, const int* __restrict__ src, const int* __restrict__ dst,
    const float* __restrict__ dinv, float* __restrict__ out, int E, int F)
{
    int wid = (blockIdx.x * 256 + threadIdx.x) >> 5;
    int lane = threadIdx.x & 31;
    if (wid >= E) return;
    int s = src[wid], d = dst[wid];
    float nrm = dinv[s] * dinv[d];
    const float* xr = xw + (size_t)s * F;
    float* orow = out + (size_t)d * F;
    for (int f = lane; f < F; f += 32)
        atomicAdd(&orow[f], xr[f] * nrm);
}
__global__ void act_kernel(float* __restrict__ p, int n, float slope) {
    int i = blockIdx.x * blockDim.x + threadIdx.x;
    if (i < n) p[i] = leakyf(p[i], slope);
}

// ---------------- GAT pieces ----------------
// wave per node: es[i] = xw[i]·a_src, ed[i] = xw[i]·a_dst (F==64)
__global__ __launch_bounds__(256) void dots_kernel(
    const float* __restrict__ xw, const float* __restrict__ as, const float* __restrict__ ad,
    float* __restrict__ es, float* __restrict__ ed, int n)
{
    int wid = (blockIdx.x * 256 + threadIdx.x) >> 5;
    int lane = threadIdx.x & 31;
    if (wid >= n) return;
    const float* row = xw + (size_t)wid * 64;
    float v0 = row[lane], v1 = row[lane + 32];
    float s = v0 * as[lane] + v1 * as[lane + 32];
    float t = v0 * ad[lane] + v1 * ad[lane + 32];
    #pragma unroll
    for (int o = 16; o > 0; o >>= 1) {
        s += __shfl_xor(s, o, 32);
        t += __shfl_xor(t, o, 32);
    }
    if (lane == 0) { es[wid] = s; ed[wid] = t; }
}
// m[i] = self-loop attention logit (init for segment max)
__global__ void gat_selfmax_kernel(const float* __restrict__ es, const float* __restrict__ ed,
                                   float* __restrict__ m, int n) {
    int i = blockIdx.x * blockDim.x + threadIdx.x;
    if (i < n) m[i] = leakyf(es[i] + ed[i], 0.2f);
}
__global__ void gat_edge_max_kernel(const float* __restrict__ es, const float* __restrict__ ed,
                                    const int* __restrict__ src, const int* __restrict__ dst,
                                    float* __restrict__ m, int E) {
    int e = blockIdx.x * blockDim.x + threadIdx.x;
    if (e >= E) return;
    atomicMaxFloat(&m[dst[e]], leakyf(es[src[e]] + ed[dst[e]], 0.2f));
}
// acc[i][f] = xw[i][f]*w_self ; dn[i] = w_self   (initializes both)
__global__ void gat_selfacc_kernel(const float* __restrict__ xw, const float* __restrict__ es,
                                   const float* __restrict__ ed, const float* __restrict__ m,
                                   float* __restrict__ acc, float* __restrict__ dn, int n) {
    int idx = blockIdx.x * blockDim.x + threadIdx.x;
    if (idx >= n * 64) return;
    int i = idx >> 6, f = idx & 63;
    float w = expf(leakyf(es[i] + ed[i], 0.2f) - m[i]);
    acc[idx] = xw[idx] * w;
    if (f == 0) dn[i] = w;
}
__global__ __launch_bounds__(256) void gat_edge_acc_kernel(
    const float* __restrict__ xw, const float* __restrict__ es, const float* __restrict__ ed,
    const int* __restrict__ src, const int* __restrict__ dst, const float* __restrict__ m,
    float* __restrict__ acc, float* __restrict__ dn, int E)
{
    int wid = (blockIdx.x * 256 + threadIdx.x) >> 5;
    int lane = threadIdx.x & 31;
    if (wid >= E) return;
    int s = src[wid], d = dst[wid];
    float w = expf(leakyf(es[s] + ed[d], 0.2f) - m[d]);
    if (lane == 0) atomicAdd(&dn[d], w);
    const float* xr = xw + (size_t)s * 64;
    float* orow = acc + (size_t)d * 64;
    #pragma unroll 2
    for (int f = lane; f < 64; f += 32)
        atomicAdd(&orow[f], xr[f] * w);
}
__global__ void gat_final_kernel(float* __restrict__ acc, const float* __restrict__ dn,
                                 const float* __restrict__ bias, int n) {
    int idx = blockIdx.x * blockDim.x + threadIdx.x;
    if (idx >= n * 64) return;
    int i = idx >> 6, f = idx & 63;
    acc[idx] = acc[idx] / dn[i] + bias[f];
}

// ---------------- pooling ----------------
__global__ void mean_pieces_kernel(const float* __restrict__ h, float* __restrict__ hc, int ngroups) {
    int idx = blockIdx.x * blockDim.x + threadIdx.x;  // ngroups*64
    if (idx >= ngroups * 64) return;
    int g = idx >> 6, f = idx & 63;
    float s = 0.f;
    const float* base = h + ((size_t)g * NPIECE) * 64 + f;
    for (int p = 0; p < NPIECE; p++) s += base[(size_t)p * 64];
    hc[idx] = s * (1.0f / NPIECE);
}
__global__ void mean_cross_kernel(const float* __restrict__ h2, float* __restrict__ hb) {
    int idx = blockIdx.x * blockDim.x + threadIdx.x;  // BATCH*64
    if (idx >= BATCH * 64) return;
    int b = idx >> 6, f = idx & 63;
    float s = 0.f;
    const float* base = h2 + ((size_t)b * NCROSS) * 64 + f;
    for (int c = 0; c < NCROSS; c++) s += base[(size_t)c * 64];
    hb[idx] = s * (1.0f / NCROSS);
}

// ---------------- cross-graph GAT: wave per destination node ----------------
// Each clique has 32 nodes <-> 32 lanes; each ordered pair occurs twice in the
// edge list (weight x2), self loop once.
__global__ __launch_bounds__(256) void gat2_kernel(
    const float* __restrict__ xw, const float* __restrict__ es, const float* __restrict__ ed,
    const float* __restrict__ bias, float* __restrict__ out, int nnodes)
{
    __shared__ float wbuf[8][32];
    int lane = threadIdx.x & 31;
    int wv = threadIdx.x >> 5;
    int d = blockIdx.x * 8 + wv;
    if (d >= nnodes) return;
    int g = d >> 5;
    int jn = (g << 5) + lane;
    float e = leakyf(es[jn] + ed[d], 0.2f);
    float m = e;
    #pragma unroll
    for (int o = 16; o > 0; o >>= 1) m = fmaxf(m, __shfl_xor(m, o, 32));
    float w = ((jn == d) ? 1.0f : 2.0f) * expf(e - m);
    float s = w;
    #pragma unroll
    for (int o = 16; o > 0; o >>= 1) s += __shfl_xor(s, o, 32);
    wbuf[wv][lane] = w / s;          // DS ops from same wave are in-order
    #pragma unroll 2
    for (int f = lane; f < 64; f += 32) {
        float a = bias[f];
        for (int j = 0; j < 32; j++)
            a += wbuf[wv][j] * xw[((size_t)((g << 5) + j)) * 64 + f];
        out[(size_t)d * 64 + f] = a;
    }
}

// ---------------- conv1d (stride 2, pad 1, k=3) + leaky + optional BN ----------------
__global__ void conv1d_kernel(const float* __restrict__ x, const float* __restrict__ W,
                              const float* __restrict__ bias, const float* __restrict__ gamma,
                              const float* __restrict__ beta, float* __restrict__ y,
                              int Bn, int Cin, int Lin, int Cout, int Lout)
{
    int idx = blockIdx.x * blockDim.x + threadIdx.x;
    if (idx >= Bn * Cout * Lout) return;
    int l = idx % Lout; int t = idx / Lout; int co = t % Cout; int b = t / Cout;
    float acc = bias[co];
    int p0 = 2 * l - 1;
    for (int ci = 0; ci < Cin; ci++) {
        const float* xr = x + ((size_t)b * Cin + ci) * Lin;
        const float* wr = W + ((size_t)co * Cin + ci) * 3;
        #pragma unroll
        for (int k = 0; k < 3; k++) {
            int p = p0 + k;
            if (p >= 0 && p < Lin) acc += xr[p] * wr[k];
        }
    }
    acc = leakyf(acc, 0.01f);
    if (gamma) acc = acc * (gamma[co] * rsqrtf(1.0f + 1e-5f)) + beta[co];
    y[idx] = acc;
}
__global__ void pool_mean_kernel(const float* __restrict__ y, float* __restrict__ p, int L) {
    int idx = blockIdx.x * blockDim.x + threadIdx.x;  // BATCH*64
    if (idx >= BATCH * 64) return;
    const float* base = y + (size_t)idx * L;
    float s = 0.f;
    for (int l = 0; l < L; l++) s += base[l];
    p[idx] = s / (float)L;
}

// ---------------- head ----------------
__global__ void head1_kernel(const float* __restrict__ hb, const float* __restrict__ pm,
                             const float* __restrict__ W, const float* __restrict__ b,
                             float* __restrict__ l2) {
    int idx = blockIdx.x * blockDim.x + threadIdx.x;  // BATCH*64
    if (idx >= BATCH * 64) return;
    int bb = idx >> 6, j = idx & 63;
    float s = b[j];
    for (int k = 0; k < 64; k++)
        s += (hb[bb * 64 + k] + pm[bb * 64 + k]) * W[k * 64 + j];
    l2[idx] = leakyf(s, 0.01f);
}
__global__ void head2_kernel(const float* __restrict__ l2, const float* __restrict__ W,
                             const float* __restrict__ b, float* __restrict__ out) {
    int bb = blockIdx.x * blockDim.x + threadIdx.x;
    if (bb >= BATCH) return;
    float s = b[0];
    for (int j = 0; j < 64; j++) s += l2[bb * 64 + j] * W[j];
    out[bb] = 1.0f / (1.0f + expf(-s));
}

// =====================================================================
extern "C" void kernel_launch(void* const* d_in, const int* in_sizes, int n_in,
                              void* d_out, int out_size, void* d_ws, size_t ws_size,
                              hipStream_t stream) {
    const float* x      = (const float*)d_in[0];
    const int*   ei     = (const int*)d_in[1];
    const float* points = (const float*)d_in[2];
    const float* W1 = (const float*)d_in[3];  const float* b1 = (const float*)d_in[4];
    const float* W2 = (const float*)d_in[5];  const float* b2 = (const float*)d_in[6];
    const float* W3 = (const float*)d_in[7];  const float* b3 = (const float*)d_in[8];
    const float* Wg1 = (const float*)d_in[9];  const float* as1 = (const float*)d_in[10];
    const float* ad1 = (const float*)d_in[11]; const float* bg1 = (const float*)d_in[12];
    const float* Wg2 = (const float*)d_in[13]; const float* as2 = (const float*)d_in[14];
    const float* ad2 = (const float*)d_in[15]; const float* bg2 = (const float*)d_in[16];
    const float* c1W = (const float*)d_in[17]; const float* c1b = (const float*)d_in[18];
    const float* g1  = (const float*)d_in[19]; const float* be1 = (const float*)d_in[20];
    const float* c2W = (const float*)d_in[21]; const float* c2b = (const float*)d_in[22];
    const float* g2  = (const float*)d_in[23]; const float* be2 = (const float*)d_in[24];
    const float* c3W = (const float*)d_in[25]; const float* c3b = (const float*)d_in[26];
    const float* Wl2 = (const float*)d_in[27]; const float* bl2 = (const float*)d_in[28];
    const float* Wl3 = (const float*)d_in[29]; const float* bl3 = (const float*)d_in[30];
    float* out = (float*)d_out;

    const int N = in_sizes[0] / 19;        // 262144
    const int E = in_sizes[1] / 2;         // 2097152
    const int* src = ei;
    const int* dst = ei + E;
    const int M = BATCH * NCROSS;          // 2048 cross-graph nodes

    // workspace layout (floats)
    float* ws = (float*)d_ws;
    const size_t NB = (size_t)N * 64;
    float* buf0 = ws;
    float* buf1 = buf0 + NB;
    float* buf2 = buf1 + NB;
    float* dinv = buf2 + NB;
    float* es   = dinv + N;
    float* ed   = es + N;
    float* mx   = ed + N;
    float* dn   = mx + N;
    // small arena aliased into buf2 (free once GAT1 GEMM has consumed h3)
    float* hc   = buf2;
    float* xw2  = hc + (size_t)M * 64;
    float* es2  = xw2 + (size_t)M * 64;
    float* ed2  = es2 + M;
    float* out2 = ed2 + M;
    float* hb   = out2 + (size_t)M * 64;
    float* pb0  = hb + BATCH * 64;                 // <= BATCH*64*1024 floats
    float* pb1  = pb0 + (size_t)BATCH * 64 * 1024;
    float* pm   = pb1 + (size_t)BATCH * 64 * 1024;
    float* l2v  = pm + BATCH * 64;

    auto cdiv = [](long a, long b) { return (int)((a + b - 1) / b); };
    const int T = 256;

    // ---- degrees: deg = 1 (self loop) + indegree; dinv = rsqrt(deg) ----
    fill_kernel<<<cdiv(N, T), T, 0, stream>>>(dinv, N, 1.0f);
    deg_kernel<<<cdiv(E, T), T, 0, stream>>>(dst, dinv, E);
    rsqrt_kernel<<<cdiv(N, T), T, 0, stream>>>(dinv, N);

    // ---- GCN layer 1: x[N,19] @ W1[19,32] -> agg -> relu, h1 = buf2 ----
    wmma_gemm_kernel<19, 32, 32><<<cdiv(N, 128), T, 0, stream>>>(x, W1, buf1);
    gcn_self_kernel<<<cdiv((long)N * 32, T), T, 0, stream>>>(buf1, dinv, b1, buf2, N, 32);
    gcn_edge_kernel<<<cdiv((long)E * 32, T), T, 0, stream>>>(buf1, src, dst, dinv, buf2, E, 32);
    act_kernel<<<cdiv((long)N * 32, T), T, 0, stream>>>(buf2, N * 32, 0.0f);

    // ---- GCN layer 2: h1 @ W2[32,64] -> agg -> leaky 0.01, h2 = buf0 ----
    wmma_gemm_kernel<32, 32, 64><<<cdiv(N, 128), T, 0, stream>>>(buf2, W2, buf1);
    gcn_self_kernel<<<cdiv((long)N * 64, T), T, 0, stream>>>(buf1, dinv, b2, buf0, N, 64);
    gcn_edge_kernel<<<cdiv((long)E * 32, T), T, 0, stream>>>(buf1, src, dst, dinv, buf0, E, 64);
    act_kernel<<<cdiv((long)N * 64, T), T, 0, stream>>>(buf0, N * 64, 0.01f);

    // ---- GCN layer 3: h2 @ W3[64,64] -> agg -> leaky 0.01, h3 = buf2 ----
    wmma_gemm_kernel<64, 64, 64><<<cdiv(N, 128), T, 0, stream>>>(buf0, W3, buf1);
    gcn_self_kernel<<<cdiv((long)N * 64, T), T, 0, stream>>>(buf1, dinv, b3, buf2, N, 64);
    gcn_edge_kernel<<<cdiv((long)E * 32, T), T, 0, stream>>>(buf1, src, dst, dinv, buf2, E, 64);
    act_kernel<<<cdiv((long)N * 64, T), T, 0, stream>>>(buf2, N * 64, 0.01f);

    // ---- GAT layer 1: xw = h3 @ Wg1 (buf1); softmax-aggregate -> buf0 ----
    wmma_gemm_kernel<64, 64, 64><<<cdiv(N, 128), T, 0, stream>>>(buf2, Wg1, buf1);
    dots_kernel<<<cdiv((long)N * 32, T), T, 0, stream>>>(buf1, as1, ad1, es, ed, N);
    gat_selfmax_kernel<<<cdiv(N, T), T, 0, stream>>>(es, ed, mx, N);
    gat_edge_max_kernel<<<cdiv(E, T), T, 0, stream>>>(es, ed, src, dst, mx, E);
    gat_selfacc_kernel<<<cdiv((long)N * 64, T), T, 0, stream>>>(buf1, es, ed, mx, buf0, dn, N);
    gat_edge_acc_kernel<<<cdiv((long)E * 32, T), T, 0, stream>>>(buf1, es, ed, src, dst, mx, buf0, dn, E);
    gat_final_kernel<<<cdiv((long)N * 64, T), T, 0, stream>>>(buf0, dn, bg1, N);

    // ---- mean over pieces: [B,C,PC,64] -> hc[B*C,64] ----
    mean_pieces_kernel<<<cdiv((long)M * 64, T), T, 0, stream>>>(buf0, hc, M);

    // ---- cross-graph GAT (2048 nodes, 32-cliques, pairs duplicated) ----
    wmma_gemm_kernel<64, 64, 64><<<cdiv(M, 128), T, 0, stream>>>(hc, Wg2, xw2);
    dots_kernel<<<cdiv((long)M * 32, T), T, 0, stream>>>(xw2, as2, ad2, es2, ed2, M);
    gat2_kernel<<<cdiv(M, 8), T, 0, stream>>>(xw2, es2, ed2, bg2, out2, M);
    mean_cross_kernel<<<cdiv(BATCH * 64, T), T, 0, stream>>>(out2, hb);

    // ---- point branch: conv1d chain ----
    conv1d_kernel<<<cdiv((long)BATCH * 32 * 1024, T), T, 0, stream>>>(
        points, c1W, c1b, g1, be1, pb0, BATCH, 3, NPTS, 32, 1024);
    conv1d_kernel<<<cdiv((long)BATCH * 64 * 512, T), T, 0, stream>>>(
        pb0, c2W, c2b, g2, be2, pb1, BATCH, 32, 1024, 64, 512);
    conv1d_kernel<<<cdiv((long)BATCH * 64 * 256, T), T, 0, stream>>>(
        pb1, c3W, c3b, nullptr, nullptr, pb0, BATCH, 64, 512, 64, 256);
    pool_mean_kernel<<<cdiv(BATCH * 64, T), T, 0, stream>>>(pb0, pm, 256);

    // ---- head ----
    head1_kernel<<<cdiv(BATCH * 64, T), T, 0, stream>>>(hb, pm, Wl2, bl2, l2v);
    head2_kernel<<<1, 64, 0, stream>>>(l2v, Wl3, bl3, out);
    (void)n_in; (void)out_size; (void)ws_size;
}
